// LinformerAttention_76012331204811
// MI455X (gfx1250) — matrix operand order repaired and checked by reading, not compiled, not run on hardware
//
#include <hip/hip_runtime.h>

typedef _Float16 f16;
typedef __attribute__((ext_vector_type(16))) _Float16 f16x16;
typedef __attribute__((ext_vector_type(8)))  float    f32x8;
typedef __attribute__((ext_vector_type(4)))  unsigned int u32x4;
typedef __attribute__((ext_vector_type(8)))  int      i32x8;
typedef __attribute__((ext_vector_type(4)))  int      i32x4;
typedef unsigned int u32;
typedef unsigned long long u64;

#define WMMA_F16(a, b, c) \
  __builtin_amdgcn_wmma_f32_16x16x32_f16(false, (a), false, (b), (short)0, (c), false, false)

// constants
#define NTOK 4096
#define BB   4
#define DIM  1024
#define NH   16
#define LRK  64
#define DH   64
#define ROWS (NTOK * BB)   // 16384

// ---------------------------------------------------------------------------
// TDM: 2D f16 tile (tile_w x tile_h elements) from global (row stride
// `stride` elements) into LDS at byte offset lds_off, with LDS padding of
// 16 bytes appended after every 64 bytes (i.e. row pitch 80B = 40 halfs).
// tile_w must be 32 (row = 64B = 16 DWORDs -> pad_interval code 3,
// pad_amount 4 DWORDs -> code 3).
// ---------------------------------------------------------------------------
__device__ __forceinline__
void tdm_load_tile_f16(const f16* gsrc, u32 lds_off,
                       u32 tensor_w, u32 tensor_h, u32 stride,
                       u32 tile_w, u32 tile_h)
{
  u64 ga = (u64)(size_t)gsrc;
  u32x4 g0;
  g0[0] = 1u;                                   // count=1, no gather
  g0[1] = lds_off;                              // LDS byte address
  g0[2] = (u32)ga;                              // global addr [31:0]
  g0[3] = (u32)(ga >> 32) | (2u << 30);         // global addr [56:32], type=2

  i32x8 g1;
  g1[0] = (int)((1u << 16)                      // data_size = 2 bytes
              | (1u << 20)                      // pad_enable
              | (3u << 22)                      // pad_interval: 16 DWORDs
              | (3u << 25));                    // pad_amount: 4 DWORDs
  g1[1] = (int)((tensor_w & 0xFFFFu) << 16);    // tensor_dim0[15:0]
  g1[2] = (int)((tensor_w >> 16) | ((tensor_h & 0xFFFFu) << 16));
  g1[3] = (int)((tensor_h >> 16) | (tile_w << 16));   // tile_dim0
  g1[4] = (int)(tile_h & 0xFFFFu);              // tile_dim1 (tile_dim2 = 0)
  g1[5] = (int)stride;                          // tensor_dim0_stride[31:0]
  g1[6] = 0;                                    // stride hi / dim1_stride lo
  g1[7] = 0;

  i32x4 z4 = {0, 0, 0, 0};
#if defined(__clang_major__) && (__clang_major__ >= 23)
  i32x8 z8 = {0, 0, 0, 0, 0, 0, 0, 0};
  __builtin_amdgcn_tensor_load_to_lds(g0, g1, z4, z4, z8, 0);
#else
  __builtin_amdgcn_tensor_load_to_lds(g0, g1, z4, z4, 0);
#endif
}

// ---------------------------------------------------------------------------
// Kernel 0: fp32 -> f16 conversion (x, W_qkv, W_out)
// ---------------------------------------------------------------------------
__global__ __launch_bounds__(256)
void k_convert(const float* __restrict__ src, f16* __restrict__ dst, int n4)
{
  int i = blockIdx.x * 256 + threadIdx.x;
  if (i < n4) {
    const float4 v = ((const float4*)src)[i];
    f16 o0 = (f16)v.x, o1 = (f16)v.y, o2 = (f16)v.z, o3 = (f16)v.w;
    f16* d = dst + (size_t)i * 4;
    d[0] = o0; d[1] = o1; d[2] = o2; d[3] = o3;
  }
}

// ---------------------------------------------------------------------------
// Kernel 1: qkv = x @ W_qkv + b_qkv -> scatter f16 q/k/v as [b*16+h][n][dh]
// A tile via TDM (double-buffered), B tile staged transposed for b128 reads.
// ---------------------------------------------------------------------------
__global__ __launch_bounds__(256)
void k_qkv_gemm(const f16* __restrict__ x16, const f16* __restrict__ W16,
                const float* __restrict__ bias,
                f16* __restrict__ q16, f16* __restrict__ k16, f16* __restrict__ v16)
{
  const int LDA = 40;    // 32 + 8 pad (halfs) -> 80B pitch (TDM pad feature)
  const int LDBT = 40;   // transposed B: [col][k], 32 + 8 pad
  __shared__ __align__(16) f16 As[2][128 * 40];
  __shared__ __align__(16) f16 BsT[128 * 40];

  const int tid  = threadIdx.x;
  const int lane = tid & 31;
  const int wave = tid >> 5;       // 0..7
  const int wm   = wave >> 2;      // 2 waves along M
  const int wn   = wave & 3;       // 4 waves along N
  const int row0 = blockIdx.y * 128;
  const int col0 = blockIdx.x * 128;

  f32x8 acc[4][2] = {};            // 64x32 per wave

  const int am  = lane & 15;
  const int akb = (lane >> 4) << 4;
  const int bn  = lane & 15;
  const int bkb = (lane >> 4) << 4;

  // prologue: kick off first A tile via TDM
  if (wave == 0)
    tdm_load_tile_f16(x16 + (size_t)row0 * DIM, (u32)(size_t)&As[0][0],
                      DIM, ROWS, DIM, 32, 128);

  for (int it = 0; it < DIM / 32; ++it) {
    const int k0 = it * 32;
    __syncthreads();   // previous compute done (Bs free, A[(it+1)&1] free)
    if (wave == 0) {
      __builtin_amdgcn_s_wait_tensorcnt((short)0);   // A[it] landed
      if (it + 1 < DIM / 32)
        tdm_load_tile_f16(x16 + (size_t)row0 * DIM + (k0 + 32),
                          (u32)(size_t)&As[(it + 1) & 1][0],
                          DIM, ROWS, DIM, 32, 128);
    }
    { // stage B transposed: BsT[col][k]
      int r = tid >> 3, cb = (tid & 7) * 16;
      const f16* src = W16 + (size_t)(k0 + r) * (3 * DIM) + col0 + cb;
      #pragma unroll
      for (int i = 0; i < 16; ++i) BsT[(cb + i) * LDBT + r] = src[i];
    }
    __syncthreads();

    const f16* Ac = As[it & 1];
    f16x16 afr[4], bfr[2];
    #pragma unroll
    for (int mt = 0; mt < 4; ++mt) {
      int rr = wm * 64 + mt * 16 + am;
      #pragma unroll
      for (int j = 0; j < 16; ++j) {
        int kk = akb + j;
        afr[mt][j] = Ac[rr * LDA + kk];
      }
    }
    #pragma unroll
    for (int nt = 0; nt < 2; ++nt) {
      int cc = wn * 32 + nt * 16 + bn;
      #pragma unroll
      for (int j = 0; j < 16; ++j)
        bfr[nt][j] = BsT[cc * LDBT + bkb + j];     // contiguous -> b128
    }
    #pragma unroll
    for (int mt = 0; mt < 4; ++mt)
      #pragma unroll
      for (int nt = 0; nt < 2; ++nt)
        acc[mt][nt] = WMMA_F16(afr[mt], bfr[nt], acc[mt][nt]);
  }

  // epilogue: `which` is uniform per block (128-col tiles don't straddle 1024)
  const int which = col0 >> 10;
  f16* dstbase = (which == 0) ? q16 : (which == 1) ? k16 : v16;
  const float scale = (which == 0) ? 0.125f : 1.0f;   // dh^-0.5 folded into q
  const int cm = (lane >> 4) << 3;
  const int cn = lane & 15;
  #pragma unroll
  for (int mt = 0; mt < 4; ++mt)
    #pragma unroll
    for (int nt = 0; nt < 2; ++nt)
      #pragma unroll
      for (int r = 0; r < 8; ++r) {
        int grow = row0 + wm * 64 + mt * 16 + cm + r;   // flat row = n*B + b
        int gcol = col0 + wn * 32 + nt * 16 + cn;
        float v = (acc[mt][nt][r] + bias[gcol]) * scale;
        int cc = gcol & 1023;
        int h = cc >> 6, d = cc & 63;
        int n = grow >> 2, b = grow & 3;
        dstbase[((size_t)(b * NH + h) * NTOK + n) * DH + d] = (f16)v;
      }
}

// ---------------------------------------------------------------------------
// Kernel 2: k_low[bh] = E^T @ k[bh],  v_low[bh] = F^T @ v[bh]  (64x64 each)
// ---------------------------------------------------------------------------
__global__ __launch_bounds__(128)
void k_lowrank(const float* __restrict__ E, const float* __restrict__ F,
               const f16* __restrict__ k16, const f16* __restrict__ v16,
               f16* __restrict__ klow, f16* __restrict__ vlow)
{
  const int bid   = blockIdx.x;
  const int which = bid & 1;
  const int bh    = bid >> 1;
  const float* Em = which ? F : E;
  const f16*   kv = (which ? v16 : k16) + (size_t)bh * NTOK * DH;
  f16*        dst = (which ? vlow : klow) + (size_t)bh * LRK * DH;

  const int tid = threadIdx.x;
  const int lane = tid & 31;
  const int wave = tid >> 5;

  __shared__ float red[LRK * DH];                // 16 KB
  for (int i = tid; i < LRK * DH; i += 128) red[i] = 0.f;
  __syncthreads();

  const int am  = lane & 15;
  const int akb = (lane >> 4) << 4;
  const int bn  = lane & 15;
  const int bkb = (lane >> 4) << 4;

  f32x8 acc[4][4] = {};

  const int nbeg = wave * (NTOK / 4);
  for (int nb = nbeg; nb < nbeg + NTOK / 4; nb += 32) {
    f16x16 afr[4], bfr[4];
    #pragma unroll
    for (int mt = 0; mt < 4; ++mt) {
      int kr = mt * 16 + am;                     // out row (low-rank index)
      #pragma unroll
      for (int j = 0; j < 16; ++j) {
        int nl = akb + j;
        afr[mt][j] = (f16)Em[(size_t)(nb + nl) * LRK + kr];   // E^T gather
      }
    }
    #pragma unroll
    for (int nt = 0; nt < 4; ++nt) {
      int d = nt * 16 + bn;
      #pragma unroll
      for (int j = 0; j < 16; ++j)
        bfr[nt][j] = kv[(size_t)(nb + bkb + j) * DH + d];
    }
    #pragma unroll
    for (int mt = 0; mt < 4; ++mt)
      #pragma unroll
      for (int nt = 0; nt < 4; ++nt)
        acc[mt][nt] = WMMA_F16(afr[mt], bfr[nt], acc[mt][nt]);
  }

  const int cm = (lane >> 4) << 3;
  const int cn = lane & 15;
  #pragma unroll
  for (int mt = 0; mt < 4; ++mt)
    #pragma unroll
    for (int nt = 0; nt < 4; ++nt)
      #pragma unroll
      for (int r = 0; r < 8; ++r)
        atomicAdd(&red[(mt * 16 + cm + r) * DH + nt * 16 + cn], acc[mt][nt][r]);
  __syncthreads();

  for (int i = tid; i < LRK * DH; i += 128) dst[i] = (f16)red[i];
}

// ---------------------------------------------------------------------------
// Kernel 3: per (bh, 64-row block): dots = q@k_low^T ; softmax ; ctx = attn@v_low
// ---------------------------------------------------------------------------
__global__ __launch_bounds__(32)
void k_attention(const f16* __restrict__ q16, const f16* __restrict__ klow,
                 const f16* __restrict__ vlow, f16* __restrict__ ctx)
{
  const int bh    = blockIdx.y;
  const int nbase = blockIdx.x * 64;
  const int lane  = threadIdx.x;

  __shared__ float sdots[64 * 64];   // 16 KB
  __shared__ f16   sattn[64 * 64];   //  8 KB

  const f16* q  = q16  + ((size_t)bh * NTOK + nbase) * DH;
  const f16* kl = klow + (size_t)bh * LRK * DH;
  const f16* vl = vlow + (size_t)bh * LRK * DH;

  const int am  = lane & 15;
  const int akb = (lane >> 4) << 4;
  const int bn  = lane & 15;
  const int bkb = (lane >> 4) << 4;
  const int cm  = (lane >> 4) << 3;
  const int cn  = lane & 15;

  // dots = q_scaled (64 x 64) @ k_low^T (64 x 64)
  f32x8 acc[4][4] = {};
  #pragma unroll
  for (int ks = 0; ks < DH; ks += 32) {
    f16x16 afr[4], bfr[4];
    #pragma unroll
    for (int mt = 0; mt < 4; ++mt)
      #pragma unroll
      for (int j = 0; j < 16; ++j) {
        int kk = akb + j;
        afr[mt][j] = q[(size_t)(mt * 16 + am) * DH + ks + kk];
      }
    #pragma unroll
    for (int nt = 0; nt < 4; ++nt)
      #pragma unroll
      for (int j = 0; j < 16; ++j) {
        int dh = ks + bkb + j;            // B row = dh (contiguous in kl row)
        int kc = nt * 16 + bn;            // B col = low-rank idx
        bfr[nt][j] = kl[(size_t)kc * DH + dh];
      }
    #pragma unroll
    for (int mt = 0; mt < 4; ++mt)
      #pragma unroll
      for (int nt = 0; nt < 4; ++nt)
        acc[mt][nt] = WMMA_F16(afr[mt], bfr[nt], acc[mt][nt]);
  }

  #pragma unroll
  for (int mt = 0; mt < 4; ++mt)
    #pragma unroll
    for (int nt = 0; nt < 4; ++nt)
      #pragma unroll
      for (int r = 0; r < 8; ++r)
        sdots[(mt * 16 + cm + r) * 64 + nt * 16 + cn] = acc[mt][nt][r];
  __syncthreads();

  // softmax over 64 cols; each lane owns 2 rows
  for (int rr = lane; rr < 64; rr += 32) {
    float* row = sdots + rr * 64;
    float mx = -1e30f;
    for (int c = 0; c < 64; ++c) mx = fmaxf(mx, row[c]);
    float sum = 0.f;
    for (int c = 0; c < 64; ++c) { float e = __expf(row[c] - mx); row[c] = e; sum += e; }
    float inv = 1.f / sum;
    for (int c = 0; c < 64; ++c) sattn[rr * 64 + c] = (f16)(row[c] * inv);
  }
  __syncthreads();

  // ctx = attn (64 x 64) @ v_low (64 x 64)
  f32x8 acc2[4][4] = {};
  #pragma unroll
  for (int ks = 0; ks < LRK; ks += 32) {
    f16x16 afr[4], bfr[4];
    #pragma unroll
    for (int mt = 0; mt < 4; ++mt)
      #pragma unroll
      for (int j = 0; j < 16; ++j) {
        int kk = akb + j;
        afr[mt][j] = sattn[(mt * 16 + am) * 64 + ks + kk];
      }
    #pragma unroll
    for (int nt = 0; nt < 4; ++nt)
      #pragma unroll
      for (int j = 0; j < 16; ++j)
        bfr[nt][j] = vl[(size_t)(ks + bkb + j) * DH + nt * 16 + bn];
    #pragma unroll
    for (int mt = 0; mt < 4; ++mt)
      #pragma unroll
      for (int nt = 0; nt < 4; ++nt)
        acc2[mt][nt] = WMMA_F16(afr[mt], bfr[nt], acc2[mt][nt]);
  }

  const int b = bh >> 4, h = bh & 15;
  #pragma unroll
  for (int mt = 0; mt < 4; ++mt)
    #pragma unroll
    for (int nt = 0; nt < 4; ++nt)
      #pragma unroll
      for (int r = 0; r < 8; ++r) {
        int n = nbase + mt * 16 + cm + r;
        int d = nt * 16 + cn;
        ctx[((size_t)n * BB + b) * DIM + h * DH + d] = (f16)acc2[mt][nt][r];
      }
}

// ---------------------------------------------------------------------------
// Kernel 4: out = ctx @ W_out + b_out (fp32 result); A via TDM, B transposed
// ---------------------------------------------------------------------------
__global__ __launch_bounds__(256)
void k_outproj(const f16* __restrict__ ctx, const f16* __restrict__ W16,
               const float* __restrict__ bias, float* __restrict__ out)
{
  const int LDA = 40;
  const int LDBT = 40;
  __shared__ __align__(16) f16 As[2][128 * 40];
  __shared__ __align__(16) f16 BsT[128 * 40];

  const int tid  = threadIdx.x;
  const int lane = tid & 31;
  const int wave = tid >> 5;
  const int wm   = wave >> 2;
  const int wn   = wave & 3;
  const int row0 = blockIdx.y * 128;
  const int col0 = blockIdx.x * 128;

  f32x8 acc[4][2] = {};
  const int am  = lane & 15;
  const int akb = (lane >> 4) << 4;
  const int bn  = lane & 15;
  const int bkb = (lane >> 4) << 4;

  if (wave == 0)
    tdm_load_tile_f16(ctx + (size_t)row0 * DIM, (u32)(size_t)&As[0][0],
                      DIM, ROWS, DIM, 32, 128);

  for (int it = 0; it < DIM / 32; ++it) {
    const int k0 = it * 32;
    __syncthreads();
    if (wave == 0) {
      __builtin_amdgcn_s_wait_tensorcnt((short)0);
      if (it + 1 < DIM / 32)
        tdm_load_tile_f16(ctx + (size_t)row0 * DIM + (k0 + 32),
                          (u32)(size_t)&As[(it + 1) & 1][0],
                          DIM, ROWS, DIM, 32, 128);
    }
    {
      int r = tid >> 3, cb = (tid & 7) * 16;
      const f16* src = W16 + (size_t)(k0 + r) * DIM + col0 + cb;
      #pragma unroll
      for (int i = 0; i < 16; ++i) BsT[(cb + i) * LDBT + r] = src[i];
    }
    __syncthreads();

    const f16* Ac = As[it & 1];
    f16x16 afr[4], bfr[2];
    #pragma unroll
    for (int mt = 0; mt < 4; ++mt) {
      int rr = wm * 64 + mt * 16 + am;
      #pragma unroll
      for (int j = 0; j < 16; ++j) {
        int kk = akb + j;
        afr[mt][j] = Ac[rr * LDA + kk];
      }
    }
    #pragma unroll
    for (int nt = 0; nt < 2; ++nt) {
      int cc = wn * 32 + nt * 16 + bn;
      #pragma unroll
      for (int j = 0; j < 16; ++j)
        bfr[nt][j] = BsT[cc * LDBT + bkb + j];
    }
    #pragma unroll
    for (int mt = 0; mt < 4; ++mt)
      #pragma unroll
      for (int nt = 0; nt < 2; ++nt)
        acc[mt][nt] = WMMA_F16(afr[mt], bfr[nt], acc[mt][nt]);
  }

  const int cm = (lane >> 4) << 3;
  const int cn = lane & 15;
  #pragma unroll
  for (int mt = 0; mt < 4; ++mt)
    #pragma unroll
    for (int nt = 0; nt < 2; ++nt)
      #pragma unroll
      for (int r = 0; r < 8; ++r) {
        int grow = row0 + wm * 64 + mt * 16 + cm + r;
        int gcol = col0 + wn * 32 + nt * 16 + cn;
        out[(size_t)grow * DIM + gcol] = acc[mt][nt][r] + bias[gcol];
      }
}

// ---------------------------------------------------------------------------
extern "C" void kernel_launch(void* const* d_in, const int* in_sizes, int n_in,
                              void* d_out, int out_size, void* d_ws, size_t ws_size,
                              hipStream_t stream) {
  (void)in_sizes; (void)n_in; (void)out_size; (void)ws_size;
  const float* x     = (const float*)d_in[0];
  const float* E     = (const float*)d_in[1];
  const float* F     = (const float*)d_in[2];
  const float* W_qkv = (const float*)d_in[3];
  const float* b_qkv = (const float*)d_in[4];
  const float* W_out = (const float*)d_in[5];
  const float* b_out = (const float*)d_in[6];
  float* out = (float*)d_out;

  char* ws = (char*)d_ws;
  const size_t SZ = (size_t)BB * NH * NTOK * DH * sizeof(f16);   // 32 MB each
  f16* q16  = (f16*)(ws);
  f16* k16  = (f16*)(ws + SZ);
  f16* v16  = (f16*)(ws + 2 * SZ);
  f16* ctx  = (f16*)(ws + 3 * SZ);
  f16* klow = (f16*)(ws + 4 * SZ);
  f16* vlow = (f16*)(ws + 4 * SZ + (size_t)BB * NH * LRK * DH * sizeof(f16));
  f16* x16  = (f16*)(ws + 4 * SZ + 2 * (size_t)BB * NH * LRK * DH * sizeof(f16));
  f16* Wq16 = x16 + (size_t)ROWS * DIM;
  f16* Wo16 = Wq16 + (size_t)DIM * 3 * DIM;

  { // fp32 -> f16 conversions
    int n4x = ROWS * DIM / 4;          // 4,194,304
    int n4q = DIM * 3 * DIM / 4;       //   786,432
    int n4o = DIM * DIM / 4;           //   262,144
    k_convert<<<(n4x + 255) / 256, 256, 0, stream>>>(x, x16, n4x);
    k_convert<<<(n4q + 255) / 256, 256, 0, stream>>>(W_qkv, Wq16, n4q);
    k_convert<<<(n4o + 255) / 256, 256, 0, stream>>>(W_out, Wo16, n4o);
  }

  dim3 g1(3 * DIM / 128, ROWS / 128);          // 24 x 128
  k_qkv_gemm<<<g1, 256, 0, stream>>>(x16, Wq16, b_qkv, q16, k16, v16);

  k_lowrank<<<BB * NH * 2, 128, 0, stream>>>(E, F, k16, v16, klow, vlow);

  dim3 g3(NTOK / 64, BB * NH);                 // 64 x 64
  k_attention<<<g3, 32, 0, stream>>>(q16, klow, vlow, ctx);

  dim3 g4(DIM / 128, ROWS / 128);              // 8 x 128
  k_outproj<<<g4, 256, 0, stream>>>(ctx, Wo16, b_out, out);
}
